// PAModuleVotes_11330123727353
// MI455X (gfx1250) — matrix operand order repaired and checked by reading, not compiled, not run on hardware
//
#include <hip/hip_runtime.h>
#include <hip/hip_bf16.h>

// ---------------------------------------------------------------------------
// Problem constants (from reference)
// ---------------------------------------------------------------------------
#define BATCH     4
#define NPTS      4096
#define C_IN      128
#define NPOINT    1024
#define NSAMPLE   32
#define N_AUG     20
#define K_ALL     52        // NSAMPLE + N_AUG
#define K_PAD     64        // padded rows per query for WMMA (4 M-tiles)
#define KCAND     64        // top-k candidates kept
#define HID       64
#define OUTC      256
#define FULL_IN   131       // C_IN + 3
#define R2        (0.4f*0.4f + 1e-6f)
#define INV_R     2.5f

// xj column layout (features-first so async B128 LDS writes are 16B aligned):
//   cols [0,128)   : gathered features
//   cols [128,131) : rel xyz
//   cols [131,160) : zero pad
#define XJ_LD     160
#define DI_NPAD   144

#define USE_ASYNC_GATHER 1

// ---------------------------------------------------------------------------
// Workspace layout (bytes)
// ---------------------------------------------------------------------------
#define WS_FPS_OFF   0                               // int  [4][1024]
#define WS_IDX_OFF   16384                           // int  [4][1024][52]
#define WS_WT_OFF    868352                          // bf16 transposed weights
// offsets inside the bf16 weight region (in ushort units)
#define D2T_OFF   0                                  // [64][64]
#define PSIT_OFF  4096                               // [160][64]   (K remapped)
#define G1T_OFF   14336                              // [64][256]
#define DIT_OFF   30720                              // [64][144]   (N remapped)
#define G2T_OFF   39936                              // [256][256]
#define AT_OFF    105472                             // [160][256]  (K remapped)
#define WT_TOTAL  146432
#define WS_FEATT_OFF (WS_WT_OFF + WT_TOTAL*2)        // bf16 [4][4096][128]

// Main-kernel LDS (per wave, bytes)
#define XJ_US     (K_PAD*XJ_LD)        // 10240
#define EN_US     (K_PAD*HID)          // 4096
#define DH_US     (K_PAD*HID)          // 4096
#define E1_US     (K_PAD*OUTC)         // 16384
#define PHI_US    64
#define WAVE_US   (XJ_US+EN_US+DH_US+E1_US+PHI_US)   // 34880
#define WAVE_BYTES (WAVE_US*2 + OUTC*4)              // +nf float[256] = 70784
#define WAVES_PER_BLOCK 2
#define MAIN_SMEM (WAVE_BYTES*WAVES_PER_BLOCK)

// ---------------------------------------------------------------------------
// Types & bf16 helpers
// ---------------------------------------------------------------------------
typedef __attribute__((ext_vector_type(16))) __bf16        v16bf;
typedef __attribute__((ext_vector_type(8)))  float         v8f;
typedef __attribute__((ext_vector_type(8)))  unsigned int  v8u;

union FragU { v8u u; v16bf b; };

__device__ __forceinline__ unsigned short f2b(float f) {
    union { float f; unsigned u; } c; c.f = f;
    unsigned u = c.u;
    u += 0x7fffu + ((u >> 16) & 1u);   // round-to-nearest-even
    return (unsigned short)(u >> 16);
}
__device__ __forceinline__ float b2f(unsigned short h) {
    union { unsigned u; float f; } c; c.u = ((unsigned)h) << 16;
    return c.f;
}
__device__ __forceinline__ v8f v8f_zero() {
    v8f z = {0.f,0.f,0.f,0.f,0.f,0.f,0.f,0.f};
    return z;
}

// ---------------------------------------------------------------------------
// 4 M-tile WMMA GEMM step: each B fragment (weight) is loaded ONCE and reused
// by 4 consecutive M-tiles (acc[0..3]); the 4 independent back-to-back WMMAs
// also cover the WMMA->WMMA RAW hazard without v_nops.
// A lane layout (ISA 7.12.2, 16-bit A 16x32): lane<16 -> M=lane, VGPR e2 holds
//   K = (e2<4 ? 2e2 : 2e2+8); lanes 16..31 add +8 to K.
// B lane layout: N = lane%16, VGPR e2 holds the same K pair as A (K, K+1).
// ---------------------------------------------------------------------------
__device__ __forceinline__ void wmma_gemm4(const unsigned short* A, int lda,
                                           const unsigned short* Bw, int ldb, int nbase,
                                           int ksteps, int lane, v8f* acc) {
    const int half = (lane >> 4) & 1;
    const int l15  = lane & 15;
    for (int ks = 0; ks < ksteps; ++ks) {
        const int kb = ks * 32;
        const unsigned short* bcol = Bw + (unsigned)kb * ldb + nbase + l15;
        FragU fb;
#pragma unroll
        for (int e2 = 0; e2 < 8; ++e2) {
            int k = ((e2 < 4) ? (2*e2) : (2*e2 + 8)) + half * 8;
            unsigned lo = bcol[k * ldb], hi = bcol[(k + 1) * ldb];
            fb.u[e2] = lo | (hi << 16);
        }
#pragma unroll
        for (int mt = 0; mt < 4; ++mt) {
            const unsigned short* arow = A + (mt*16 + l15) * lda;
            FragU fa;
#pragma unroll
            for (int e2 = 0; e2 < 8; ++e2) {
                int k = kb + ((e2 < 4) ? (2*e2) : (2*e2 + 8)) + half * 8;
                unsigned lo = arow[k], hi = arow[k + 1];
                fa.u[e2] = lo | (hi << 16);
            }
            acc[mt] = __builtin_amdgcn_wmma_f32_16x16x32_bf16(false, fa.b, false, fb.b,
                                                              (short)0, acc[mt], false, false);
        }
    }
}

// ---------------------------------------------------------------------------
// Prep: transpose+convert weight (O,I) f32 -> bf16 W^T [Kp][Np], zero padded.
// remapK/remapN: map padded xj-layout index -> reference input index
//   (c<128 -> feat -> i=c+3 ; 128<=c<131 -> rel -> i=c-128 ; else pad)
// ---------------------------------------------------------------------------
__global__ void k_wtrans(const float* __restrict__ w, unsigned short* __restrict__ dst,
                         int O, int I, int Kp, int Np, int remapK, int remapN) {
    int i = blockIdx.x * blockDim.x + threadIdx.x;
    int tot = Kp * Np;
    if (i >= tot) return;
    int k = i / Np, n = i % Np;
    int sk = k, sn = n;
    if (remapK) sk = (k < 128) ? (k + 3) : (k < FULL_IN ? (k - 128) : -1);
    if (remapN) sn = (n < 128) ? (n + 3) : (n < FULL_IN ? (n - 128) : -1);
    float v = (sk >= 0 && sk < I && sn >= 0 && sn < O) ? w[sn * I + sk] : 0.f;
    dst[i] = f2b(v);
}

// Prep: features [B][C][N] f32 -> bf16 [B][N][C]
__global__ void k_ftrans(const float* __restrict__ f, unsigned short* __restrict__ dst) {
    int i = blockIdx.x * blockDim.x + threadIdx.x;
    if (i >= BATCH * NPTS * C_IN) return;
    int c = i % C_IN;
    int n = (i / C_IN) % NPTS;
    int b = i / (C_IN * NPTS);
    dst[i] = f2b(f[(b * C_IN + c) * NPTS + n]);
}

// ---------------------------------------------------------------------------
// FPS: one block per batch, sequential farthest point sampling
// ---------------------------------------------------------------------------
__global__ __launch_bounds__(256) void k_fps(const float* __restrict__ xyz,
                                             int* __restrict__ fps_ws,
                                             float* __restrict__ out_newxyz,
                                             int* __restrict__ out_inds) {
    __shared__ float sx[NPTS * 3];
    __shared__ float rv[256];
    __shared__ int   ri[256];
    __shared__ float cp[3];
    const int b = blockIdx.x, tid = threadIdx.x;
    for (int i = tid; i < NPTS * 3; i += 256) sx[i] = xyz[b * NPTS * 3 + i];
    __syncthreads();

    float dist[NPTS / 256];
    const float x0 = sx[0], y0 = sx[1], z0 = sx[2];
#pragma unroll
    for (int t = 0; t < NPTS / 256; ++t) {
        int n = tid + 256 * t;
        float dx = sx[n*3] - x0, dy = sx[n*3+1] - y0, dz = sx[n*3+2] - z0;
        dist[t] = dx*dx + dy*dy + dz*dz;
    }
    if (tid == 0) {
        fps_ws[b * NPOINT] = 0;
        out_inds[b * NPOINT] = 0;
        out_newxyz[(b * NPOINT) * 3 + 0] = x0;
        out_newxyz[(b * NPOINT) * 3 + 1] = y0;
        out_newxyz[(b * NPOINT) * 3 + 2] = z0;
    }
    for (int it = 1; it < NPOINT; ++it) {
        float bv = -1.f; int bn = 0x7fffffff;
#pragma unroll
        for (int t = 0; t < NPTS / 256; ++t) {
            int n = tid + 256 * t;
            float v = dist[t];
            if (v > bv || (v == bv && n < bn)) { bv = v; bn = n; }
        }
        rv[tid] = bv; ri[tid] = bn;
        __syncthreads();
        for (int s = 128; s > 0; s >>= 1) {
            if (tid < s) {
                float ov = rv[tid + s]; int on = ri[tid + s];
                if (ov > rv[tid] || (ov == rv[tid] && on < ri[tid])) { rv[tid] = ov; ri[tid] = on; }
            }
            __syncthreads();
        }
        if (tid == 0) {
            int sel = ri[0];
            cp[0] = sx[sel*3]; cp[1] = sx[sel*3+1]; cp[2] = sx[sel*3+2];
            fps_ws[b * NPOINT + it] = sel;
            out_inds[b * NPOINT + it] = sel;
            out_newxyz[(b * NPOINT + it) * 3 + 0] = cp[0];
            out_newxyz[(b * NPOINT + it) * 3 + 1] = cp[1];
            out_newxyz[(b * NPOINT + it) * 3 + 2] = cp[2];
        }
        __syncthreads();
        float px = cp[0], py = cp[1], pz = cp[2];
#pragma unroll
        for (int t = 0; t < NPTS / 256; ++t) {
            int n = tid + 256 * t;
            float dx = sx[n*3] - px, dy = sx[n*3+1] - py, dz = sx[n*3+2] - pz;
            float d = dx*dx + dy*dy + dz*dz;
            dist[t] = fminf(dist[t], d);
        }
        __syncthreads();
    }
}

// ---------------------------------------------------------------------------
// KNN top-64 + ball/aug selection. One thread per query (insertion list).
// Distance-sorted candidates collapse the masked top_k logic to index math.
// ---------------------------------------------------------------------------
__global__ __launch_bounds__(64) void k_knn(const float* __restrict__ xyz,
                                            const float* __restrict__ newxyz,
                                            int* __restrict__ idx_ws) {
    __shared__ float sx[NPTS * 3];
    const int q = blockIdx.x * 64 + threadIdx.x;      // 0..4095
    const int b = q >> 10;
    for (int i = threadIdx.x; i < NPTS * 3; i += 64) sx[i] = xyz[b * NPTS * 3 + i];
    __syncthreads();

    const float px = newxyz[q*3], py = newxyz[q*3+1], pz = newxyz[q*3+2];
    float bd[KCAND]; int bi[KCAND];
#pragma unroll
    for (int i = 0; i < KCAND; ++i) { bd[i] = 3.0e38f; bi[i] = 0; }

    for (int n = 0; n < NPTS; ++n) {
        float dx = sx[n*3] - px, dy = sx[n*3+1] - py, dz = sx[n*3+2] - pz;
        float d = dx*dx + dy*dy + dz*dz;
        if (d < bd[KCAND-1]) {
            int pos = KCAND - 1;
            while (pos > 0 && bd[pos-1] > d) {
                bd[pos] = bd[pos-1]; bi[pos] = bi[pos-1]; --pos;
            }
            bd[pos] = d; bi[pos] = n;
        }
    }
    int m = 0;
    for (int i = 0; i < KCAND; ++i) if (bd[i] <= R2) ++m;
    int m2 = m < K_ALL ? m : K_ALL;

    int* out = idx_ws + q * K_ALL;
    for (int j = 0; j < NSAMPLE; ++j) out[j] = (j < m) ? bi[j] : bi[0];
    for (int j = 0; j < N_AUG; ++j) {
        int t = m2 + j;
        out[NSAMPLE + j] = (t < K_ALL) ? bi[t] : bi[0];
    }
}

// ---------------------------------------------------------------------------
// Fused attention MLP: one wave32 per query, all GEMMs via WMMA bf16.
// ---------------------------------------------------------------------------
struct MainParams {
    const float* xyz;
    const unsigned short* featT;
    const float* newxyz;
    const int* idxAll;
    const int* fpsInds;
    const unsigned short* wT;
    const float *phi_w, *phi_s, *phi_b;
    const float *psi_s, *psi_b;
    const float *d1_w, *d1_s, *d1_b;
    const float *d2_s, *d2_b;
    const float *di_s, *di_b;
    const float *g1_s, *g1_b;
    const float *g2_s, *g2_b;
    const float *a_s,  *a_b;
    const float *o_w,  *o_s, *o_b;
    float* outNF;
};

__global__ __launch_bounds__(32 * WAVES_PER_BLOCK) void k_main(MainParams P) {
    extern __shared__ unsigned char smem[];
    const int wid  = threadIdx.x >> 5;
    const int lane = threadIdx.x & 31;
    const int half = lane >> 4;
    const int l15  = lane & 15;
    const int q    = blockIdx.x * WAVES_PER_BLOCK + wid;   // 0..4095
    const int b    = q >> 10;
    const int qi   = q & 1023;

    unsigned char* base = smem + wid * WAVE_BYTES;
    unsigned short* XJ  = (unsigned short*)base;
    unsigned short* EN  = XJ + XJ_US;
    unsigned short* DH  = EN + EN_US;
    unsigned short* E1  = DH + DH_US;
    unsigned short* PHI = E1 + E1_US;
    float*          NF  = (float*)(base + WAVE_US * 2);

    if (lane == 0) {   // warm the big L2-resident weight blocks
        __builtin_prefetch(P.wT + G2T_OFF, 0, 3);
        __builtin_prefetch(P.wT + AT_OFF, 0, 3);
    }

    // ---------------- gather xj = [feats | rel | pad] ------------------------
    const float px = P.newxyz[q*3], py = P.newxyz[q*3+1], pz = P.newxyz[q*3+2];
    const int* myidx = P.idxAll + q * K_ALL;

#if USE_ASYNC_GATHER
    // Async LDS DMA: 2 rows / instruction (lanes 0-15 row k, 16-31 row k+1),
    // 16B per lane, ASYNCcnt-tracked (GLOBAL_LOAD_ASYNC_TO_LDS_B128).
    for (int it = 0; it < K_ALL / 2; ++it) {
        int kk = it * 2 + half;
        int j = myidx[kk];
        const unsigned short* g = P.featT + (unsigned)(b * NPTS + j) * C_IN + l15 * 8;
        unsigned lds = (unsigned)(wid * WAVE_BYTES + kk * (XJ_LD * 2) + l15 * 16);
        unsigned long long ga = (unsigned long long)g;
        asm volatile("global_load_async_to_lds_b128 %0, %1, off"
                     :: "v"(lds), "v"(ga) : "memory");
    }
#else
    for (int k = 0; k < K_ALL; ++k) {
        int j = myidx[k];
        const unsigned short* fr = P.featT + (unsigned)(b * NPTS + j) * C_IN;
#pragma unroll
        for (int t = 0; t < 4; ++t) {
            int c = lane * 4 + t;
            XJ[k * XJ_LD + c] = fr[c];
        }
    }
#endif
    for (int k = 0; k < K_ALL; ++k) {               // rel + zero pad columns
        int j = myidx[k];
        if (lane < 3) {
            float gx = P.xyz[(b * NPTS + j) * 3 + lane];
            float pc = (lane == 0) ? px : (lane == 1 ? py : pz);
            XJ[k * XJ_LD + 128 + lane] = f2b((gx - pc) * INV_R);
        }
        if (lane < XJ_LD - FULL_IN)
            XJ[k * XJ_LD + FULL_IN + lane] = 0;
    }
    for (int i = lane; i < (K_PAD - K_ALL) * XJ_LD; i += 32)   // zero rows 52..63
        XJ[K_ALL * XJ_LD + i] = 0;
#if USE_ASYNC_GATHER
    asm volatile("s_wait_asynccnt 0" ::: "memory");
#endif

    // ---------------- phi = relu(cbr(xi)) (single row, VALU) -----------------
    {
        int fi = P.fpsInds[q];
        const unsigned short* fr = P.featT + (unsigned)(b * NPTS + fi) * C_IN;
#pragma unroll
        for (int t = 0; t < 4; ++t) { int c = lane * 4 + t; NF[c] = b2f(fr[c]); }
#pragma unroll
        for (int t = 0; t < 2; ++t) {
            int n = lane * 2 + t;
            float acc = 0.f;
            const float* wr = P.phi_w + n * FULL_IN + 3;
            for (int c = 0; c < C_IN; ++c) acc += NF[c] * wr[c];
            PHI[n] = f2b(fmaxf(acc * P.phi_s[n] + P.phi_b[n], 0.f));
        }
    }

    // ---------------- dh1 = relu(cbr(rel, d1)) -> EN (VALU, K=3) -------------
    for (int t = 0; t < (K_PAD * HID) / 32; ++t) {
        int id = lane + 32 * t;
        int k = id >> 6, n = id & 63;
        float r0 = b2f(XJ[k * XJ_LD + 128]);
        float r1 = b2f(XJ[k * XJ_LD + 129]);
        float r2v = b2f(XJ[k * XJ_LD + 130]);
        float acc = r0 * P.d1_w[n*3] + r1 * P.d1_w[n*3+1] + r2v * P.d1_w[n*3+2];
        EN[k * HID + n] = f2b(fmaxf(acc * P.d1_s[n] + P.d1_b[n], 0.f));
    }

    // ---------------- dh = relu(cbr(dh1, d2)) : WMMA 64x64x64 ----------------
    for (int nt = 0; nt < 4; ++nt) {
        v8f acc[4] = {v8f_zero(), v8f_zero(), v8f_zero(), v8f_zero()};
        wmma_gemm4(EN, HID, P.wT + D2T_OFF, HID, nt*16, 2, lane, acc);
        int n = nt*16 + l15;
        float s = P.d2_s[n], bb = P.d2_b[n];
#pragma unroll
        for (int mt = 0; mt < 4; ++mt)
#pragma unroll
            for (int v = 0; v < 8; ++v) {
                int m = mt*16 + half*8 + v;
                DH[m * HID + n] = f2b(fmaxf(acc[mt][v] * s + bb, 0.f));
            }
    }

    // ---------------- psi + energy = phi - psi + dh : WMMA 64x64x160 ---------
    for (int nt = 0; nt < 4; ++nt) {
        v8f acc[4] = {v8f_zero(), v8f_zero(), v8f_zero(), v8f_zero()};
        wmma_gemm4(XJ, XJ_LD, P.wT + PSIT_OFF, HID, nt*16, 5, lane, acc);
        int n = nt*16 + l15;
        float s = P.psi_s[n], bb = P.psi_b[n], ph = b2f(PHI[n]);
#pragma unroll
        for (int mt = 0; mt < 4; ++mt)
#pragma unroll
            for (int v = 0; v < 8; ++v) {
                int m = mt*16 + half*8 + v;
                float psi = fmaxf(acc[mt][v] * s + bb, 0.f);
                EN[m * HID + n] = f2b(ph - psi + b2f(DH[m * HID + n]));
            }
    }

    // ---------------- e1 = relu(cbr(energy, g1)) : WMMA 64x256x64 ------------
    for (int nt = 0; nt < 16; ++nt) {
        v8f acc[4] = {v8f_zero(), v8f_zero(), v8f_zero(), v8f_zero()};
        wmma_gemm4(EN, HID, P.wT + G1T_OFF, OUTC, nt*16, 2, lane, acc);
        int n = nt*16 + l15;
        float s = P.g1_s[n], bb = P.g1_b[n];
#pragma unroll
        for (int mt = 0; mt < 4; ++mt)
#pragma unroll
            for (int v = 0; v < 8; ++v) {
                int m = mt*16 + half*8 + v;
                E1[m * OUTC + n] = f2b(fmaxf(acc[mt][v] * s + bb, 0.f));
            }
    }

    // ---------------- vd = xj + cbr(dh, di) : WMMA 64x144x64, in-place XJ ----
    for (int nt = 0; nt < DI_NPAD / 16; ++nt) {
        v8f acc[4] = {v8f_zero(), v8f_zero(), v8f_zero(), v8f_zero()};
        wmma_gemm4(DH, HID, P.wT + DIT_OFF, DI_NPAD, nt*16, 2, lane, acc);
        int n = nt*16 + l15;                     // xj column index
        if (n < FULL_IN) {
            int sn = (n < 128) ? (n + 3) : (n - 128);   // reference output index
            float s = P.di_s[sn], bb = P.di_b[sn];
#pragma unroll
            for (int mt = 0; mt < 4; ++mt)
#pragma unroll
                for (int v = 0; v < 8; ++v) {
                    int m = mt*16 + half*8 + v;
                    float xv = b2f(XJ[m * XJ_LD + n]);
                    XJ[m * XJ_LD + n] = f2b(xv + acc[mt][v] * s + bb);
                }
        }
    }

    // ---------------- fused g2 logits + a values + masked softmax ------------
    for (int nt = 0; nt < 16; ++nt) {
        int nb = nt * 16;
        int n = nb + l15;
        v8f lg[4] = {v8f_zero(), v8f_zero(), v8f_zero(), v8f_zero()};
        v8f va[4] = {v8f_zero(), v8f_zero(), v8f_zero(), v8f_zero()};
        wmma_gemm4(E1, OUTC, P.wT + G2T_OFF, OUTC, nb, 8, lane, lg);
        wmma_gemm4(XJ, XJ_LD, P.wT + AT_OFF,  OUTC, nb, 5, lane, va);

        float gs = P.g2_s[n], gb = P.g2_b[n];
        float as_ = P.a_s[n], ab = P.a_b[n];
        float mx = -3.0e38f;
#pragma unroll
        for (int mt = 0; mt < 4; ++mt)
#pragma unroll
            for (int v = 0; v < 8; ++v) {
                int m = mt*16 + half*8 + v;
                if (m < K_ALL) mx = fmaxf(mx, lg[mt][v] * gs + gb);
            }
        mx = fmaxf(mx, __shfl_xor(mx, 16, 32));
        float num = 0.f, den = 0.f;
#pragma unroll
        for (int mt = 0; mt < 4; ++mt)
#pragma unroll
            for (int v = 0; v < 8; ++v) {
                int m = mt*16 + half*8 + v;
                if (m < K_ALL) {
                    float l = lg[mt][v] * gs + gb;
                    float p = __expf(l - mx);
                    float vv = fmaxf(va[mt][v] * as_ + ab, 0.f);
                    num += p * vv;
                    den += p;
                }
            }
        num += __shfl_xor(num, 16, 32);
        den += __shfl_xor(den, 16, 32);
        if (lane < 16) NF[n] = num / den;
    }

    // ---------------- out = relu(cbr(nf, o)), write transposed ---------------
#pragma unroll
    for (int t = 0; t < 8; ++t) {
        int n = lane + 32 * t;
        float acc = 0.f;
        const float* wr = P.o_w + n * OUTC;
        for (int c = 0; c < OUTC; ++c) acc += NF[c] * wr[c];
        float y = fmaxf(acc * P.o_s[n] + P.o_b[n], 0.f);
        P.outNF[(b * OUTC + n) * NPOINT + qi] = y;
    }
}

// ---------------------------------------------------------------------------
// Host launcher
// ---------------------------------------------------------------------------
extern "C" void kernel_launch(void* const* d_in, const int* in_sizes, int n_in,
                              void* d_out, int out_size, void* d_ws, size_t ws_size,
                              hipStream_t stream) {
    const float* xyz      = (const float*)d_in[0];
    const float* features = (const float*)d_in[1];
    const float* phi_w = (const float*)d_in[2];
    const float* phi_s = (const float*)d_in[3];
    const float* phi_b = (const float*)d_in[4];
    const float* psi_w = (const float*)d_in[5];
    const float* psi_s = (const float*)d_in[6];
    const float* psi_b = (const float*)d_in[7];
    const float* d1_w  = (const float*)d_in[8];
    const float* d1_s  = (const float*)d_in[9];
    const float* d1_b  = (const float*)d_in[10];
    const float* d2_w  = (const float*)d_in[11];
    const float* d2_s  = (const float*)d_in[12];
    const float* d2_b  = (const float*)d_in[13];
    const float* di_w  = (const float*)d_in[14];
    const float* di_s  = (const float*)d_in[15];
    const float* di_b  = (const float*)d_in[16];
    const float* g1_w  = (const float*)d_in[17];
    const float* g1_s  = (const float*)d_in[18];
    const float* g1_b  = (const float*)d_in[19];
    const float* g2_w  = (const float*)d_in[20];
    const float* g2_s  = (const float*)d_in[21];
    const float* g2_b  = (const float*)d_in[22];
    const float* a_w   = (const float*)d_in[23];
    const float* a_s   = (const float*)d_in[24];
    const float* a_b   = (const float*)d_in[25];
    const float* o_w   = (const float*)d_in[26];
    const float* o_s   = (const float*)d_in[27];
    const float* o_b   = (const float*)d_in[28];

    // outputs: new_xyz [4,1024,3] f32, nf [4,256,1024] f32, fps_inds [4,1024] int
    float* out_newxyz = (float*)d_out;
    float* out_nf     = out_newxyz + BATCH * NPOINT * 3;
    int*   out_inds   = (int*)(out_nf + BATCH * OUTC * NPOINT);

    unsigned char* ws = (unsigned char*)d_ws;
    int* fps_ws = (int*)(ws + WS_FPS_OFF);
    int* idx_ws = (int*)(ws + WS_IDX_OFF);
    unsigned short* wT    = (unsigned short*)(ws + WS_WT_OFF);
    unsigned short* featT = (unsigned short*)(ws + WS_FEATT_OFF);

    // --- prep: transposed bf16 weights + transposed bf16 features ---
    auto launch_wt = [&](const float* w, int off, int O, int I, int Kp, int Np,
                         int rK, int rN) {
        int tot = Kp * Np;
        k_wtrans<<<(tot + 255) / 256, 256, 0, stream>>>(w, wT + off, O, I, Kp, Np, rK, rN);
    };
    launch_wt(d2_w, D2T_OFF, HID,  HID,     HID, HID,     0, 0);
    launch_wt(psi_w, PSIT_OFF, HID, FULL_IN, XJ_LD, HID,  1, 0);
    launch_wt(g1_w, G1T_OFF, OUTC, HID,     HID, OUTC,    0, 0);
    launch_wt(di_w, DIT_OFF, FULL_IN, HID,  HID, DI_NPAD, 0, 1);
    launch_wt(g2_w, G2T_OFF, OUTC, OUTC,    OUTC, OUTC,   0, 0);
    launch_wt(a_w,  AT_OFF,  OUTC, FULL_IN, XJ_LD, OUTC,  1, 0);

    {
        int tot = BATCH * NPTS * C_IN;
        k_ftrans<<<(tot + 255) / 256, 256, 0, stream>>>(features, featT);
    }

    // --- FPS ---
    k_fps<<<BATCH, 256, 0, stream>>>(xyz, fps_ws, out_newxyz, out_inds);

    // --- KNN + ball/aug selection ---
    k_knn<<<(BATCH * NPOINT) / 64, 64, 0, stream>>>(xyz, out_newxyz, idx_ws);

    // --- fused WMMA attention MLP ---
    (void)hipFuncSetAttribute((const void*)k_main,
                              hipFuncAttributeMaxDynamicSharedMemorySize, MAIN_SMEM);
    MainParams P;
    P.xyz = xyz; P.featT = featT; P.newxyz = out_newxyz;
    P.idxAll = idx_ws; P.fpsInds = fps_ws; P.wT = wT;
    P.phi_w = phi_w; P.phi_s = phi_s; P.phi_b = phi_b;
    P.psi_s = psi_s; P.psi_b = psi_b;
    P.d1_w = d1_w; P.d1_s = d1_s; P.d1_b = d1_b;
    P.d2_s = d2_s; P.d2_b = d2_b;
    P.di_s = di_s; P.di_b = di_b;
    P.g1_s = g1_s; P.g1_b = g1_b;
    P.g2_s = g2_s; P.g2_b = g2_b;
    P.a_s = a_s; P.a_b = a_b;
    P.o_w = o_w; P.o_s = o_s; P.o_b = o_b;
    P.outNF = out_nf;
    k_main<<<(BATCH * NPOINT) / WAVES_PER_BLOCK, 32 * WAVES_PER_BLOCK, MAIN_SMEM, stream>>>(P);
}
